// SpectralPooling_4475355923020
// MI455X (gfx1250) — matrix operand order repaired and checked by reading, not compiled, not run on hardware
//
#include <hip/hip_runtime.h>

// SpectralPooling on gfx1250 (MI455X), fully fused, bf16x3 error-compensated WMMA.
//
// Math: dctn over all 5 axes + crop/pad on last 3 + inverse dctn. The orthonormal
// DCTs on axes 0-1 cancel exactly, leaving 256 independent 64^3 volumes, each a
// separable 3-axis contraction with one combined 32x64 matrix
//   A = M32[:28,:]^T @ M64[:28,:]   (DCT64 -> crop 28 -> pad 32 -> iDCT32).
//
// Roofline: 256MB in + 32MB out = 12.4us at 23.3 TB/s.  bf16x3 WMMA compute
// (3x 7.5 GFLOP effective) ~9us => memory bound, fp32-grade accuracy.
// One workgroup (8 wave32) per volume; everything else lives in LDS (~196KB).

typedef __bf16 bf16_t;
typedef __attribute__((ext_vector_type(16))) __bf16 bf16x16;
typedef __attribute__((ext_vector_type(8)))  __bf16 bf16x8;
typedef __attribute__((ext_vector_type(4)))  __bf16 bf16x4;
typedef __attribute__((ext_vector_type(2)))  __bf16 bf16x2;
typedef __attribute__((ext_vector_type(8)))  float  f32x8;
typedef __attribute__((ext_vector_type(4)))  float  f32x4;

__device__ __forceinline__ f32x8 wmma_bf16(bf16x16 a, bf16x16 b, f32x8 c) {
  return __builtin_amdgcn_wmma_f32_16x16x32_bf16(false, a, false, b,
                                                 (short)0, c, false, false);
}

// A/B fragment from two contiguous 16B chunks (ds_load_b128 / global_load_b128).
__device__ __forceinline__ bf16x16 frag2(const bf16_t* p0, const bf16_t* p1) {
  bf16x8 a = *(const bf16x8*)p0;
  bf16x8 b = *(const bf16x8*)p1;
  return __builtin_shufflevector(a, b, 0,1,2,3,4,5,6,7,8,9,10,11,12,13,14,15);
}

// Error-compensated split: f ~= hi + lo, both bf16 (RNE).
__device__ __forceinline__ void split2(float f, __bf16& h, __bf16& l) {
  h = (__bf16)f;
  l = (__bf16)(f - (float)h);
}

// 3-term compensated product accumulate: acc += (ah+al)*(bh+bl) - O(2^-16)
__device__ __forceinline__ f32x8 wmma3(bf16x16 ah, bf16x16 al,
                                       bf16x16 bh, bf16x16 bl, f32x8 acc) {
  acc = wmma_bf16(ah, bh, acc);
  acc = wmma_bf16(ah, bl, acc);
  acc = wmma_bf16(al, bh, acc);
  return acc;
}

// ---------------- init: A[j,n] = sum_{k<28} M32[k,j]*M64[k,n], split hi/lo ----
__global__ __launch_bounds__(256)
void k_init(bf16_t* __restrict__ Ah, bf16_t* __restrict__ Al) {
  int e = blockIdx.x * 256 + threadIdx.x;   // grid 8 -> 2048 entries
  if (e >= 2048) return;
  int j = e >> 6, n = e & 63;
  const double PI = 3.14159265358979323846;
  double s = 0.0;
  for (int k = 0; k < 28; ++k) {
    double s32 = (k == 0) ? 0.17677669529663689 : 0.25;               // sqrt(1/32), sqrt(2/32)
    double s64 = (k == 0) ? 0.125               : 0.17677669529663689; // sqrt(1/64), sqrt(2/64)
    double m32 = s32 * cos(PI * (2.0 * j + 1.0) * (double)k / 64.0);
    double m64 = s64 * cos(PI * (2.0 * n + 1.0) * (double)k / 128.0);
    s += m32 * m64;
  }
  __bf16 h, l;
  split2((float)s, h, l);
  Ah[e] = h; Al[e] = l;
}

// ---------------- fused kernel: one block per 64^3 volume --------------------
__global__ __launch_bounds__(256)
void k_fused(const float* __restrict__ x, const bf16_t* __restrict__ Awh,
             const bf16_t* __restrict__ Awl, float* __restrict__ out) {
  __shared__ bf16_t Ash[32 * 72], Asl[32 * 72];   // A hi/lo, stride 72
  __shared__ bf16_t Xsh[64 * 72], Xsl[64 * 72];   // X slice hi/lo
  __shared__ bf16_t Wth[32 * 72], Wtl[32 * 72];   // W^T = (X A^T)^T : [j2][n1]
  __shared__ bf16_t Tbh[1024 * 40], Tbl[1024 * 40]; // T^T: [m=j1*32+j2][n0 local]

  const int t    = threadIdx.x;
  const int v    = blockIdx.x;
  const int w    = t >> 5;
  const int lane = t & 31, l15 = lane & 15, hi = lane >> 4;

  { // stage A hi/lo into LDS (one 16B chunk of each per thread)
    int e = t * 8;
    int row = e >> 6, col = e & 63;
    *(bf16x8*)&Ash[row * 72 + col] = *(const bf16x8*)&Awh[e];
    *(bf16x8*)&Asl[row * 72 + col] = *(const bf16x8*)&Awl[e];
  }

  const float* xv = x + (size_t)v * 262144;
  f32x8 acc3[8][2] = {};     // stage-3 accumulators: 8 m-tiles x 2 j0-tiles per wave

  // prefetch slice 0 into registers (non-temporal: X is a read-once 256MB stream)
  f32x4 fr[4];
  #pragma unroll
  for (int i = 0; i < 4; ++i)
    fr[i] = __builtin_nontemporal_load((const f32x4*)xv + (t + 256 * i));

  for (int pass = 0; pass < 2; ++pass) {
    for (int n0l = 0; n0l < 32; ++n0l) {
      const int s = pass * 32 + n0l;
      __syncthreads();   // Xs free of step-1 readers; As ready on iter 0

      // deposit prefetched slice as hi/lo bf16 into LDS
      #pragma unroll
      for (int i = 0; i < 4; ++i) {
        int c = t + 256 * i;
        int row = c >> 4, col = (c & 15) * 4;
        bf16x4 hv, lv;
        #pragma unroll
        for (int q = 0; q < 4; ++q) { __bf16 h, l; split2(fr[i][q], h, l); hv[q] = h; lv[q] = l; }
        *(bf16x4*)&Xsh[row * 72 + col] = hv;
        *(bf16x4*)&Xsl[row * 72 + col] = lv;
      }
      __syncthreads();

      // prefetch next slice (overlaps steps 1-3; waited at next deposit)
      if (s + 1 < 64) {
        const f32x4* xs4 = (const f32x4*)(xv + (s + 1) * 4096);
        #pragma unroll
        for (int i = 0; i < 4; ++i)
          fr[i] = __builtin_nontemporal_load(xs4 + (t + 256 * i));
      }

      // step 1: W[n1,j2] = sum_n2 X[n1,n2] A[j2,n2]; 8 waves = 4x2 tiles
      {
        const int Mt = w >> 1, Nt = w & 1;
        const int arow = (Mt * 16 + l15) * 72, brow = (Nt * 16 + l15) * 72;
        f32x8 acc = {};
        #pragma unroll
        for (int k0 = 0; k0 < 64; k0 += 32) {
          int ab = k0 + 8 * hi, bb = k0 + 8 * hi;
          bf16x16 ah = frag2(&Xsh[arow + ab], &Xsh[arow + ab + 16]);
          bf16x16 al = frag2(&Xsl[arow + ab], &Xsl[arow + ab + 16]);
          bf16x16 bh = frag2(&Ash[brow + bb], &Ash[brow + bb + 16]);
          bf16x16 bl = frag2(&Asl[brow + bb], &Asl[brow + bb + 16]);
          acc = wmma3(ah, al, bh, bl, acc);
        }
        // store transposed hi/lo, packed pairs (n1 = Mt*16 + 8*hi + 2r + {0,1})
        #pragma unroll
        for (int r = 0; r < 4; ++r) {
          __bf16 h0, l0, h1, l1;
          split2(acc[2 * r], h0, l0);
          split2(acc[2 * r + 1], h1, l1);
          bf16x2 ph, pl; ph[0] = h0; ph[1] = h1; pl[0] = l0; pl[1] = l1;
          int off = brow + Mt * 16 + 8 * hi + 2 * r;
          *(bf16x2*)&Wth[off] = ph;
          *(bf16x2*)&Wtl[off] = pl;
        }
      }
      __syncthreads();

      // step 2: T[j1,j2] = sum_n1 A[j1,n1] W[n1,j2]; waves 0..3 = 2x2 tiles
      if (w < 4) {
        const int Mt = w >> 1, Nt = w & 1;
        const int arow = (Mt * 16 + l15) * 72, brow = (Nt * 16 + l15) * 72;
        f32x8 acc = {};
        #pragma unroll
        for (int k0 = 0; k0 < 64; k0 += 32) {
          int ab = k0 + 8 * hi, bb = k0 + 8 * hi;
          bf16x16 ah = frag2(&Ash[arow + ab], &Ash[arow + ab + 16]);
          bf16x16 al = frag2(&Asl[arow + ab], &Asl[arow + ab + 16]);
          bf16x16 bh = frag2(&Wth[brow + bb], &Wth[brow + bb + 16]);
          bf16x16 bl = frag2(&Wtl[brow + bb], &Wtl[brow + bb + 16]);
          acc = wmma3(ah, al, bh, bl, acc);
        }
        #pragma unroll
        for (int r = 0; r < 8; ++r) {
          int m = (Mt * 16 + r + 8 * hi) * 32 + Nt * 16 + l15;
          __bf16 h, l; split2(acc[r], h, l);
          Tbh[m * 40 + n0l] = h;     // 20-bank lane stride: conflict-free
          Tbl[m * 40 + n0l] = l;
        }
      }
    }
    __syncthreads();

    // step 3: out[j0,m] += sum_{n0 in pass} A[j0,n0] T[n0,m]; all 8 waves,
    // K-window = this pass's 32 columns of Tb; B = rows of A at kb = pass*32.
    {
      const int kb = pass * 32 + 8 * hi;
      bf16x16 bh0 = frag2(&Ash[l15 * 72 + kb],        &Ash[l15 * 72 + kb + 16]);
      bf16x16 bl0 = frag2(&Asl[l15 * 72 + kb],        &Asl[l15 * 72 + kb + 16]);
      bf16x16 bh1 = frag2(&Ash[(16 + l15) * 72 + kb], &Ash[(16 + l15) * 72 + kb + 16]);
      bf16x16 bl1 = frag2(&Asl[(16 + l15) * 72 + kb], &Asl[(16 + l15) * 72 + kb + 16]);
      const int ab = 8 * hi;
      #pragma unroll
      for (int i = 0; i < 8; ++i) {
        int m = ((w * 8 + i) * 16 + l15) * 40;
        bf16x16 ah = frag2(&Tbh[m + ab], &Tbh[m + ab + 16]);
        bf16x16 al = frag2(&Tbl[m + ab], &Tbl[m + ab + 16]);
        acc3[i][0] = wmma3(ah, al, bh0, bl0, acc3[i][0]);
        acc3[i][1] = wmma3(ah, al, bh1, bl1, acc3[i][1]);
      }
    }
    __syncthreads();   // Tb consumed before next pass's step 2 overwrites it
  }

  // final store: D VGPR r -> m = Mt*16 + 8*hi + r (contiguous -> float4 pairs)
  float* ov = out + (size_t)v * 32768;
  #pragma unroll
  for (int i = 0; i < 8; ++i) {
    const int mbase = (w * 8 + i) * 16 + 8 * hi;
    #pragma unroll
    for (int nt = 0; nt < 2; ++nt) {
      const int j0 = nt * 16 + l15;
      f32x8 a = acc3[i][nt];
      f32x4 lo = {a[0], a[1], a[2], a[3]};
      f32x4 hv = {a[4], a[5], a[6], a[7]};
      *(f32x4*)&ov[j0 * 1024 + mbase]     = lo;
      *(f32x4*)&ov[j0 * 1024 + mbase + 4] = hv;
    }
  }
}

extern "C" void kernel_launch(void* const* d_in, const int* in_sizes, int n_in,
                              void* d_out, int out_size, void* d_ws, size_t ws_size,
                              hipStream_t stream) {
  (void)in_sizes; (void)n_in; (void)out_size; (void)ws_size;
  const float* x = (const float*)d_in[0];
  float* out = (float*)d_out;
  bf16_t* Ah = (bf16_t*)d_ws;                       // 2048 bf16
  bf16_t* Al = (bf16_t*)((char*)d_ws + 4096);       // 2048 bf16

  k_init <<<8,   256, 0, stream>>>(Ah, Al);
  k_fused<<<256, 256, 0, stream>>>(x, Ah, Al, out); // one block per volume
}
